// LocallyGroupedSelfAttention_79860621902359
// MI455X (gfx1250) — matrix operand order repaired and checked by reading, not compiled, not run on hardware
//
#include <hip/hip_runtime.h>

// ---------------------------------------------------------------------------
// CDNA5 (gfx1250) WMMA types
// ---------------------------------------------------------------------------
typedef __attribute__((ext_vector_type(16))) __bf16 v16bf;
typedef __attribute__((ext_vector_type(8)))  float  v8f;
typedef __attribute__((ext_vector_type(4)))  int    v4i;

// Problem constants (from reference setup_inputs)
#define BB      32
#define HH      56
#define WW      56
#define NN      (HH * WW)      // 3136
#define CC      512
#define HEADS   8
#define HD      64             // CC / HEADS
#define WS      7
#define SS      49             // WS*WS tokens per window
#define GG      64             // 8x8 windows
#define MTOK    (BB * NN)      // 100352 tokens

// ---------------------------------------------------------------------------
// Async global -> LDS (16B per lane).  gfx1250 async-DMA path
// (GLOBAL_LOAD_ASYNC_TO_LDS_B128, ASYNCcnt-tracked); builtin signature takes
// addrspace(1) int4* src and addrspace(3) int4* dst.
// ---------------------------------------------------------------------------
typedef v4i __attribute__((address_space(1)))* gbl_v4i_ptr;
typedef v4i __attribute__((address_space(3)))* lds_v4i_ptr;

#if __has_builtin(__builtin_amdgcn_global_load_async_to_lds_b128)
#define ASYNC_LDS_PATH 1
__device__ inline void copy16_to_lds(__bf16* lds_dst, const __bf16* gsrc) {
  __builtin_amdgcn_global_load_async_to_lds_b128(
      (gbl_v4i_ptr)(gsrc), (lds_v4i_ptr)(lds_dst), 0, 0);
}
#else
#define ASYNC_LDS_PATH 0
__device__ inline void copy16_to_lds(__bf16* lds_dst, const __bf16* gsrc) {
  *(uint4*)lds_dst = *(const uint4*)gsrc;
}
#endif

__device__ inline void wait_async_lds() {
#if __has_builtin(__builtin_amdgcn_s_wait_asynccnt)
  __builtin_amdgcn_s_wait_asynccnt(0);
#else
  asm volatile("s_wait_asynccnt 0" ::: "memory");
#endif
}

// ---------------------------------------------------------------------------
// WMMA fragment loaders (LDS row-major, bf16, arbitrary element stride)
// Contiguous k order: element s of lane half h holds k = 16h + s, same in A and B (wave32).
// ---------------------------------------------------------------------------

// A matrix 16x32 (M x K), row-major with element stride `stride`.
__device__ inline v16bf load_a_frag(const __bf16* base, int stride, int lane) {
  int row  = lane & 15;
  int half = lane >> 4;
  v16bf a;
#pragma unroll
  for (int i = 0; i < 8; ++i) {
    int k = half * 16 + 2 * i;
    a[2 * i]     = base[row * stride + k];
    a[2 * i + 1] = base[row * stride + k + 1];
  }
  return a;
}

// B matrix 32x16 (K x N), row-major with element stride `stride`.
__device__ inline v16bf load_b_frag(const __bf16* base, int stride, int lane) {
  int col  = lane & 15;
  int half = lane >> 4;
  v16bf b;
#pragma unroll
  for (int i = 0; i < 8; ++i) {
    int k = half * 16 + 2 * i;
    b[2 * i]     = base[k * stride + col];
    b[2 * i + 1] = base[(k + 1) * stride + col];
  }
  return b;
}

// B matrix 32x16 where the underlying storage is the TRANSPOSE ([n][k]).
__device__ inline v16bf load_bT_frag(const __bf16* base, int stride, int lane) {
  int col  = lane & 15;   // n
  int half = lane >> 4;
  v16bf b;
#pragma unroll
  for (int i = 0; i < 8; ++i) {
    int k = half * 16 + 2 * i;
    b[2 * i]     = base[col * stride + k];
    b[2 * i + 1] = base[col * stride + k + 1];
  }
  return b;
}

// ---------------------------------------------------------------------------
// f32 -> bf16 conversion (x4 vectorized)
// ---------------------------------------------------------------------------
struct alignas(8) bf16x4 { __bf16 a, b, c, d; };

__global__ __launch_bounds__(256) void cvt_f32_to_bf16(
    const float* __restrict__ in, __bf16* __restrict__ out, int n4) {
  int i = blockIdx.x * 256 + threadIdx.x;
  if (i < n4) {
    float4 v = ((const float4*)in)[i];
    bf16x4 o;
    o.a = (__bf16)v.x; o.b = (__bf16)v.y; o.c = (__bf16)v.z; o.d = (__bf16)v.w;
    ((bf16x4*)out)[i] = o;
  }
}

// ---------------------------------------------------------------------------
// Tiled bf16 GEMM:  C[M,N] = A[M,K] * B[K,N]  (f32 accumulate)
// BM=128 BN=128 BK=32, 256 threads = 8 waves, each wave owns a 64x32 subtile
// (4 M-tiles x 2 N-tiles of 16x16).  Double-buffered LDS with async DMA:
// next tile streams into buf^1 while WMMAs consume buf.
// ---------------------------------------------------------------------------
template <bool OUT_BF16>
__global__ __launch_bounds__(256) void gemm_bf16_wmma(
    const __bf16* __restrict__ A, const __bf16* __restrict__ B,
    __bf16* __restrict__ outb, float* __restrict__ outf,
    const float* __restrict__ bias, int M, int N, int K) {
  constexpr int BM = 128, BK = 32;
  constexpr int ASTR = 40;    // 32 + 8 pad, 80B rows (16B aligned)
  constexpr int BSTR = 136;   // 128 + 8 pad, 272B rows (16B aligned)
  __shared__ alignas(16) __bf16 As[2][BM * ASTR];
  __shared__ alignas(16) __bf16 Bs[2][BK * BSTR];

  const int tid  = threadIdx.x;
  const int lane = tid & 31;
  const int w    = tid >> 5;     // 0..7
  const int wm   = w & 1;        // 2 wave-rows of 64
  const int wn   = w >> 1;       // 4 wave-cols of 32
  const long m_block = (long)blockIdx.x * BM;
  const long n_block = (long)blockIdx.y * 128;

  const int rowA = tid >> 2, chA = tid & 3;   // A: 128x32, 2 chunks/thread
  const int rowB = tid >> 3, chB = tid & 7;   // B: 32x128, 2 chunks/thread

  v8f acc[4][2] = {};

  auto issue_tile = [&](int k0, int buf) {
    copy16_to_lds(&As[buf][rowA * ASTR + chA * 8],
                  A + (m_block + rowA) * K + k0 + chA * 8);
    copy16_to_lds(&As[buf][(rowA + 64) * ASTR + chA * 8],
                  A + (m_block + rowA + 64) * K + k0 + chA * 8);
    copy16_to_lds(&Bs[buf][rowB * BSTR + chB * 8],
                  B + (long)(k0 + rowB) * N + n_block + chB * 8);
    copy16_to_lds(&Bs[buf][rowB * BSTR + (chB + 8) * 8],
                  B + (long)(k0 + rowB) * N + n_block + (chB + 8) * 8);
  };

  issue_tile(0, 0);
  wait_async_lds();
  __syncthreads();

  const int KT = K / BK;
  for (int kt = 0; kt < KT; ++kt) {
    const int cur = kt & 1;
    if (kt + 1 < KT) issue_tile((kt + 1) * BK, cur ^ 1);   // overlap DMA w/ math

    const __bf16* Ab = &As[cur][0];
    const __bf16* Bb = &Bs[cur][0];
    v16bf bfr[2];
#pragma unroll
    for (int ni = 0; ni < 2; ++ni)
      bfr[ni] = load_b_frag(Bb + wn * 32 + ni * 16, BSTR, lane);
#pragma unroll
    for (int mi = 0; mi < 4; ++mi) {
      v16bf af = load_a_frag(Ab + (wm * 64 + mi * 16) * ASTR, ASTR, lane);
#pragma unroll
      for (int ni = 0; ni < 2; ++ni)
        acc[mi][ni] = __builtin_amdgcn_wmma_f32_16x16x32_bf16(
            false, af, false, bfr[ni], (short)0, acc[mi][ni], false, false);
    }
    wait_async_lds();
    __syncthreads();
  }

  // --- epilogue: C/D layout = 8 f32, M = (lane>>4)*8 + r, N = lane&15
  const int n_l = lane & 15;
  const int mb  = (lane >> 4) * 8;
#pragma unroll
  for (int mi = 0; mi < 4; ++mi) {
#pragma unroll
    for (int ni = 0; ni < 2; ++ni) {
      long m0 = m_block + wm * 64 + mi * 16 + mb;
      long c  = n_block + wn * 32 + ni * 16 + n_l;
#pragma unroll
      for (int r = 0; r < 8; ++r) {
        if (OUT_BF16) {
          outb[(m0 + r) * N + c] = (__bf16)acc[mi][ni][r];
        } else {
          outf[(m0 + r) * N + c] = acc[mi][ni][r] + bias[c];
        }
      }
    }
  }
}

// ---------------------------------------------------------------------------
// Windowed attention: one wave per (b, g, head).
// Q,K,V tiles 49x64 zero-padded to 64x64 in LDS (bf16),
// scores 64x64 f32 in LDS, softmax over 49 cols, P*V via WMMA.
// ---------------------------------------------------------------------------
#define QSTR 72   // 64 + 8 pad -> 144B rows (16B aligned)
#define SSTR 66   // f32 stride, breaks bank conflicts
#define PSTR 72

__global__ __launch_bounds__(32) void window_attn_wmma(
    const __bf16* __restrict__ qkv, __bf16* __restrict__ aout) {
  __shared__ alignas(16) __bf16 Qs[64 * QSTR];
  __shared__ alignas(16) __bf16 Ks[64 * QSTR];
  __shared__ alignas(16) __bf16 Vs[64 * QSTR];
  __shared__ alignas(16) __bf16 Ps[64 * PSTR];
  __shared__ float Ss[64 * SSTR];

  const int blk  = blockIdx.x;
  const int h    = blk & 7;
  const int g    = (blk >> 3) & 63;
  const int b    = blk >> 9;
  const int lane = threadIdx.x;

  const int grow = (g >> 3) * WS;   // window origin in image
  const int gcol = (g & 7) * WS;

  // ---- zero-fill pad rows 49..63 (15 rows x 8 chunks), then async-gather
  {
    const uint4 zero = make_uint4(0u, 0u, 0u, 0u);
    for (int i = lane; i < 15 * 8; i += 32) {
      int s  = SS + i / 8;
      int ch = i % 8;
      *(uint4*)&Qs[s * QSTR + ch * 8] = zero;
      *(uint4*)&Ks[s * QSTR + ch * 8] = zero;
      *(uint4*)&Vs[s * QSTR + ch * 8] = zero;
    }
#pragma unroll
    for (int it = 0; it < 13; ++it) {        // rows 0..51 (last iter partial)
      int s  = it * 4 + (lane >> 3);         // 0..51
      int ch = lane & 7;                     // 8 bf16 per chunk
      if (s < SS) {
        int rimg = grow + s / WS;
        int cimg = gcol + s % WS;
        long t = (long)b * NN + rimg * WW + cimg;
        const __bf16* base = qkv + t * (3 * CC) + h * HD + ch * 8;
        copy16_to_lds(&Qs[s * QSTR + ch * 8], base);
        copy16_to_lds(&Ks[s * QSTR + ch * 8], base + CC);
        copy16_to_lds(&Vs[s * QSTR + ch * 8], base + 2 * CC);
      }
    }
  }
  wait_async_lds();
  __syncthreads();

  const int n_l = lane & 15;
  const int mb  = (lane >> 4) * 8;

  // ---- S = scale * Q @ K^T   (64x64, K-dim = 64 = 2 x 32)
#pragma unroll
  for (int mi = 0; mi < 4; ++mi) {
#pragma unroll
    for (int ni = 0; ni < 4; ++ni) {
      v8f acc = {};
#pragma unroll
      for (int kk = 0; kk < 2; ++kk) {
        v16bf a  = load_a_frag(Qs + mi * 16 * QSTR + kk * 32, QSTR, lane);
        v16bf bt = load_bT_frag(Ks + ni * 16 * QSTR + kk * 32, QSTR, lane);
        acc = __builtin_amdgcn_wmma_f32_16x16x32_bf16(
            false, a, false, bt, (short)0, acc, false, false);
      }
#pragma unroll
      for (int r = 0; r < 8; ++r)
        Ss[(mi * 16 + mb + r) * SSTR + ni * 16 + n_l] = acc[r] * 0.125f; // hd^-0.5
    }
  }
  __syncthreads();

  // ---- softmax over cols 0..48, write P (bf16), pad cols -> 0
#pragma unroll
  for (int rr = 0; rr < 2; ++rr) {
    int row = lane + rr * 32;
    float mx = -1e30f;
    for (int c = 0; c < SS; ++c) mx = fmaxf(mx, Ss[row * SSTR + c]);
    float sum = 0.f;
    for (int c = 0; c < SS; ++c) {
      float e = __expf(Ss[row * SSTR + c] - mx);
      Ss[row * SSTR + c] = e;
      sum += e;
    }
    float inv = 1.f / sum;
#pragma unroll 1
    for (int c = 0; c < 64; ++c) {
      float p = (c < SS) ? Ss[row * SSTR + c] * inv : 0.f;
      Ps[row * PSTR + c] = (__bf16)p;
    }
  }
  __syncthreads();

  // ---- O = P @ V  (64x64 x 64x64), store valid rows back to token layout
#pragma unroll
  for (int mi = 0; mi < 4; ++mi) {
#pragma unroll
    for (int ni = 0; ni < 4; ++ni) {
      v8f acc = {};
#pragma unroll
      for (int kk = 0; kk < 2; ++kk) {
        v16bf a  = load_a_frag(Ps + mi * 16 * PSTR + kk * 32, PSTR, lane);
        v16bf bf = load_b_frag(Vs + (kk * 32) * QSTR + ni * 16, QSTR, lane);
        acc = __builtin_amdgcn_wmma_f32_16x16x32_bf16(
            false, a, false, bf, (short)0, acc, false, false);
      }
#pragma unroll
      for (int r = 0; r < 8; ++r) {
        int s = mi * 16 + mb + r;
        if (s < SS) {
          int rimg = grow + s / WS;
          int cimg = gcol + s % WS;
          long t = (long)b * NN + rimg * WW + cimg;
          aout[t * CC + h * HD + ni * 16 + n_l] = (__bf16)acc[r];
        }
      }
    }
  }
}

// ---------------------------------------------------------------------------
// Launcher
// ---------------------------------------------------------------------------
extern "C" void kernel_launch(void* const* d_in, const int* in_sizes, int n_in,
                              void* d_out, int out_size, void* d_ws, size_t ws_size,
                              hipStream_t stream) {
  const float* x      = (const float*)d_in[0];
  const float* w_qkv  = (const float*)d_in[1];
  const float* w_proj = (const float*)d_in[2];
  const float* b_proj = (const float*)d_in[3];

  char* ws = (char*)d_ws;
  const size_t off_xb   = 0;                                     // 100352*512*2
  const size_t off_qkv  = off_xb  + (size_t)MTOK * CC * 2;       // 100352*1536*2
  const size_t off_attn = off_qkv + (size_t)MTOK * 3 * CC * 2;   // 100352*512*2
  const size_t off_w1   = off_attn + (size_t)MTOK * CC * 2;      // 512*1536*2
  const size_t off_w2   = off_w1  + (size_t)CC * 3 * CC * 2;     // 512*512*2

  __bf16* xb    = (__bf16*)(ws + off_xb);
  __bf16* qkvb  = (__bf16*)(ws + off_qkv);
  __bf16* attnb = (__bf16*)(ws + off_attn);
  __bf16* w1b   = (__bf16*)(ws + off_w1);
  __bf16* w2b   = (__bf16*)(ws + off_w2);

  // 1) convert inputs to bf16
  {
    int n4 = (MTOK * CC) / 4;
    cvt_f32_to_bf16<<<(n4 + 255) / 256, 256, 0, stream>>>(x, xb, n4);
  }
  {
    int n4 = (CC * 3 * CC) / 4;
    cvt_f32_to_bf16<<<(n4 + 255) / 256, 256, 0, stream>>>(w_qkv, w1b, n4);
  }
  {
    int n4 = (CC * CC) / 4;
    cvt_f32_to_bf16<<<(n4 + 255) / 256, 256, 0, stream>>>(w_proj, w2b, n4);
  }

  // 2) QKV GEMM: [100352,512] x [512,1536] -> bf16
  gemm_bf16_wmma<true><<<dim3(MTOK / 128, (3 * CC) / 128), 256, 0, stream>>>(
      xb, w1b, qkvb, nullptr, nullptr, MTOK, 3 * CC, CC);

  // 3) windowed attention: one wave per (b, g, head)
  window_attn_wmma<<<BB * GG * HEADS, 32, 0, stream>>>(qkvb, attnb);

  // 4) output projection + bias: [100352,512] x [512,512] -> f32
  gemm_bf16_wmma<false><<<dim3(MTOK / 128, CC / 128), 256, 0, stream>>>(
      attnb, w2b, nullptr, (float*)d_out, b_proj, MTOK, CC, CC);
}
